// CrfRnnLayer_43344809951906
// MI455X (gfx1250) — matrix operand order repaired and checked by reading, not compiled, not run on hardware
//
#include <hip/hip_runtime.h>
#include <hip/hip_bf16.h>

// ---- problem constants -----------------------------------------------------
#define HH 80
#define WW 80
#define CC 21
#define NN 6400            // HH*WW
#define CP 32              // C padded to 2 WMMA M-tiles
#define NUM_ITER 5
#define NSTRIP 4           // 16-col strips per wave (64 output cols/wave)
#define KSPLIT 4           // K-dimension split factor
#define KCHUNK (NN / KSPLIT)

typedef __attribute__((ext_vector_type(16))) __bf16 v16bf;
typedef __attribute__((ext_vector_type(8)))  __bf16 v8bf;
typedef __attribute__((ext_vector_type(8)))  float  v8f;

// ---------------------------------------------------------------------------
// 1) Build bf16 Gaussian kernel matrices Ks (spatial) and Kb (bilateral).
//    K[i*N+j] = exp(-0.5*||f_i - f_j||^2).  Symmetric.
// ---------------------------------------------------------------------------
__global__ void crf_build_kernels(const float* __restrict__ rgb,
                                  __bf16* __restrict__ Ks,
                                  __bf16* __restrict__ Kb) {
    const int j = blockIdx.x * 16 + (threadIdx.x & 15);
    const int i = blockIdx.y * 16 + (threadIdx.x >> 4);
    const float xi = (float)(i % WW), yi = (float)(i / WW);
    const float xj = (float)(j % WW), yj = (float)(j / WW);
    const float dx = xi - xj, dy = yi - yj;
    const float dp2 = dx * dx + dy * dy;

    // spatial: pos / 3
    const float ds2 = dp2 * (1.0f / 9.0f);
    Ks[(size_t)i * NN + j] = (__bf16)__expf(-0.5f * ds2);

    // bilateral: pos/160, rgb/3
    float db2 = dp2 * (1.0f / (160.0f * 160.0f));
    #pragma unroll
    for (int c = 0; c < 3; ++c) {
        const float dc = rgb[i * 3 + c] - rgb[j * 3 + c];
        db2 += dc * dc * (1.0f / 9.0f);
    }
    Kb[(size_t)i * NN + j] = (__bf16)__expf(-0.5f * db2);
}

// ---------------------------------------------------------------------------
// 2) rnorm[row] = 1 / sum_i K[row][i]   (symmetric: row sum == column sum)
// ---------------------------------------------------------------------------
__global__ void crf_row_norm(const __bf16* __restrict__ K,
                             float* __restrict__ rnorm) {
    __shared__ float red[256];
    const int row = blockIdx.x;
    const __bf16* p = K + (size_t)row * NN;
    float s = 0.0f;
    for (int c = threadIdx.x * 8; c < NN; c += 256 * 8) {
        v8bf v = *(const v8bf*)(p + c);
        #pragma unroll
        for (int t = 0; t < 8; ++t) s += (float)v[t];
    }
    red[threadIdx.x] = s;
    __syncthreads();
    for (int off = 128; off > 0; off >>= 1) {
        if ((int)threadIdx.x < off) red[threadIdx.x] += red[threadIdx.x + off];
        __syncthreads();
    }
    if (threadIdx.x == 0) rnorm[row] = 1.0f / red[0];
}

// ---------------------------------------------------------------------------
// 3) Collapse the per-iteration CxC chain once:
//    Ms = compat @ spatial_kernel,  Mb = compat @ bilateral_kernel
// ---------------------------------------------------------------------------
__global__ void crf_mixmat(const float* __restrict__ Sk,
                           const float* __restrict__ Bk,
                           const float* __restrict__ Cm,
                           float* __restrict__ Ms,
                           float* __restrict__ Mb) {
    const int t = threadIdx.x;
    if (t >= CC * CC) return;
    const int r = t / CC, c = t % CC;
    float s = 0.0f, b = 0.0f;
    for (int k = 0; k < CC; ++k) {
        const float cm = Cm[r * CC + k];
        s += cm * Sk[k * CC + c];
        b += cm * Bk[k * CC + c];
    }
    Ms[t] = s;
    Mb[t] = b;
}

// ---------------------------------------------------------------------------
// 4) Transpose unaries (N,C) -> (C,N) into both u_t (constant) and q (state).
// ---------------------------------------------------------------------------
__global__ void crf_init_q(const float* __restrict__ unaries,
                           float* __restrict__ ut,
                           float* __restrict__ q) {
    const int t = blockIdx.x * blockDim.x + threadIdx.x;
    if (t >= CC * NN) return;
    const int c = t / NN, j = t % NN;
    const float v = unaries[j * CC + c];
    ut[t] = v;
    q[t]  = v;
}

// ---------------------------------------------------------------------------
// 5) Column softmax over classes: q (C,N) f32 -> sm (32,N) bf16, rows 21..31=0
// ---------------------------------------------------------------------------
__global__ void crf_softmax(const float* __restrict__ q,
                            __bf16* __restrict__ sm) {
    const int j = blockIdx.x * blockDim.x + threadIdx.x;
    if (j >= NN) return;
    float v[CC];
    float m = -3.4e38f;
    #pragma unroll
    for (int c = 0; c < CC; ++c) {
        v[c] = q[c * NN + j];
        m = fmaxf(m, v[c]);
    }
    float s = 0.0f;
    #pragma unroll
    for (int c = 0; c < CC; ++c) {
        v[c] = __expf(v[c] - m);
        s += v[c];
    }
    const float inv = 1.0f / s;
    #pragma unroll
    for (int c = 0; c < CC; ++c) sm[c * NN + j] = (__bf16)(v[c] * inv);
    #pragma unroll
    for (int c = CC; c < CP; ++c) sm[c * NN + j] = (__bf16)0.0f;
}

// ---------------------------------------------------------------------------
// 6) Core WMMA GEMM:  part = sm @ K  (partial over one K chunk)
//    One wave owns 4 adjacent 16-col strips (64 cols), both M tiles, and one
//    K chunk of 1600.  K is SYMMETRIC so B[k][n] = K[j0+n][i0+k] loads as a
//    contiguous 32B row chunk per lane; the 4 strips' B tiles use constant
//    immediate byte offsets (s*204800) off a single address register pair.
//    Branch-free software pipeline: each fragment buffer is consumed by its
//    8-WMMA group and then refilled in place (WAR forces issue order, no
//    rotation copies possible), so every wait is a partial wait with a full
//    WMMA group of latency hiding behind it.
//    Wave-job decode: mat = w&1, ksplit = (w>>1)&3, colgroup = w>>3.
// ---------------------------------------------------------------------------
__global__ void __launch_bounds__(256)
crf_gemm(const __bf16* __restrict__ sm,
         const __bf16* __restrict__ Ks, const __bf16* __restrict__ Kb,
         float* __restrict__ part /* [2][KSPLIT][CP][NN] f32 */) {
    const int lane = threadIdx.x & 31;
    const int w = blockIdx.x * (blockDim.x >> 5) + (threadIdx.x >> 5); // 0..799
    const int mat = w & 1;
    const int ksp = (w >> 1) & (KSPLIT - 1);
    const int grp = w >> 3;                        // 0..99
    const __bf16* __restrict__ K = mat ? Kb : Ks;
    const int j0 = grp * (16 * NSTRIP);
    const int k0 = ksp * KCHUNK;

    // B fragment (wave32): column n = lane&15; lanes 0-15 hold K=0..15,
    // lanes 16-31 hold K=16..31 — contiguous 32 bytes per lane (row-major,
    // via symmetry K[i][j] == K[j][i]).
    const __bf16* bptr = K + (size_t)(j0 + (lane & 15)) * NN
                           + ((lane >> 4) * 16) + k0;

    // A fragment: row m = lane&15, group g = lane>>4;
// VGPR0-7 hold k = g*16..g*16+15 (contiguous, matching B)
    const int aRow = lane & 15;
    const int aOff = (lane >> 4) * 16;
    const __bf16* aptr0 = sm + (size_t)aRow * NN + aOff + k0;          // M 0-15
    const __bf16* aptr1 = sm + (size_t)(aRow + 16) * NN + aOff + k0;   // M 16-31

    v8f acc[NSTRIP][2] = {};

    auto loadfrag = [&](int k, v16bf* b, v16bf& a0, v16bf& a1) {
        #pragma unroll
        for (int s = 0; s < NSTRIP; ++s)
            b[s] = *(const v16bf*)(bptr + (size_t)s * 16 * NN + k);
        const v8bf l0 = *(const v8bf*)(aptr0 + k);
        const v8bf h0 = *(const v8bf*)(aptr0 + k + 8);
        const v8bf l1 = *(const v8bf*)(aptr1 + k);
        const v8bf h1 = *(const v8bf*)(aptr1 + k + 8);
        a0 = __builtin_shufflevector(l0, h0,
            0,1,2,3,4,5,6,7,8,9,10,11,12,13,14,15);
        a1 = __builtin_shufflevector(l1, h1,
            0,1,2,3,4,5,6,7,8,9,10,11,12,13,14,15);
    };
    auto mmacc = [&](const v16bf* b, const v16bf& a0, const v16bf& a1) {
        #pragma unroll
        for (int s = 0; s < NSTRIP; ++s) {
            acc[s][0] = __builtin_amdgcn_wmma_f32_16x16x32_bf16(
                false, a0, false, b[s], (short)0, acc[s][0], false, false);
            acc[s][1] = __builtin_amdgcn_wmma_f32_16x16x32_bf16(
                false, a1, false, b[s], (short)0, acc[s][1], false, false);
        }
    };

    v16bf bA[NSTRIP], bB[NSTRIP];
    v16bf a0A, a1A, a0B, a1B;
    loadfrag(0,  bA, a0A, a1A);
    loadfrag(32, bB, a0B, a1B);
    int k = 0;
    // KCHUNK/32 = 50 k-steps -> 24 steady-state pairs + epilogue pair
    #pragma unroll 1
    for (; k + 64 < KCHUNK; k += 64) {
        mmacc(bA, a0A, a1A);                 // consume k
        loadfrag(k + 64, bA, a0A, a1A);      // refill in place (k+64)
        mmacc(bB, a0B, a1B);                 // consume k+32
        loadfrag(k + 96, bB, a0B, a1B);      // refill in place (k+96)
    }
    mmacc(bA, a0A, a1A);                     // k = KCHUNK-64
    mmacc(bB, a0B, a1B);                     // k = KCHUNK-32

    // Epilogue: C/D layout — VGPR r holds M = r + 8*(lane>=16), N = lane&15.
    float* __restrict__ pout = part + (size_t)(mat * KSPLIT + ksp) * (CP * NN);
    const int n = lane & 15;
    const int mb = (lane >> 4) * 8;
    #pragma unroll
    for (int s = 0; s < NSTRIP; ++s) {
        const int jc = j0 + s * 16 + n;
        #pragma unroll
        for (int r = 0; r < 8; ++r) {
            pout[(size_t)(mb + r)      * NN + jc] = acc[s][0][r];
            pout[(size_t)(16 + mb + r) * NN + jc] = acc[s][1][r];
        }
    }
}

// ---------------------------------------------------------------------------
// 7) Combine K-split partials and apply 1/norm column scaling (rows 0..20).
// ---------------------------------------------------------------------------
__global__ void crf_reduce(const float* __restrict__ part,
                           const float* __restrict__ rns,
                           const float* __restrict__ rnb,
                           float* __restrict__ so,
                           float* __restrict__ bo) {
    const int t = blockIdx.x * blockDim.x + threadIdx.x;
    if (t >= CC * NN) return;
    const int j = t % NN;
    const int c = t / NN;
    const size_t idx = (size_t)c * NN + j;
    const size_t M = (size_t)CP * NN;
    float s = 0.0f, b = 0.0f;
    #pragma unroll
    for (int ksp = 0; ksp < KSPLIT; ++ksp) {
        s += part[(size_t)(0 * KSPLIT + ksp) * M + idx];
        b += part[(size_t)(1 * KSPLIT + ksp) * M + idx];
    }
    so[t] = s * rns[j];
    bo[t] = b * rnb[j];
}

// ---------------------------------------------------------------------------
// 8) q = u_t - Ms @ so - Mb @ bo   (per column; Ms/Mb reads are scalar-uniform)
// ---------------------------------------------------------------------------
__global__ void crf_update_q(const float* __restrict__ ut,
                             const float* __restrict__ so,
                             const float* __restrict__ bo,
                             const float* __restrict__ Ms,
                             const float* __restrict__ Mb,
                             float* __restrict__ q) {
    const int j = blockIdx.x * blockDim.x + threadIdx.x;
    const int c = blockIdx.y;
    if (j >= NN) return;
    float acc = ut[c * NN + j];
    #pragma unroll
    for (int k = 0; k < CC; ++k) {
        acc -= Ms[c * CC + k] * so[k * NN + j];
        acc -= Mb[c * CC + k] * bo[k * NN + j];
    }
    q[c * NN + j] = acc;
}

// ---------------------------------------------------------------------------
// 9) (C,N) -> (1,H,W,C) output transpose
// ---------------------------------------------------------------------------
__global__ void crf_write_out(const float* __restrict__ q,
                              float* __restrict__ out) {
    const int t = blockIdx.x * blockDim.x + threadIdx.x;
    if (t >= NN * CC) return;
    const int j = t / CC, c = t % CC;
    out[t] = q[c * NN + j];
}

// ---------------------------------------------------------------------------
extern "C" void kernel_launch(void* const* d_in, const int* in_sizes, int n_in,
                              void* d_out, int out_size, void* d_ws, size_t ws_size,
                              hipStream_t stream) {
    const float* unaries = (const float*)d_in[0];
    const float* rgb     = (const float*)d_in[1];
    const float* Sk      = (const float*)d_in[2];
    const float* Bk      = (const float*)d_in[3];
    const float* Cm      = (const float*)d_in[4];
    float* out = (float*)d_out;

    // workspace carve-out (256B aligned)
    char* ws = (char*)d_ws;
    size_t off = 0;
    auto carve = [&](size_t bytes) -> void* {
        void* p = ws + off;
        off = (off + bytes + 255) & ~(size_t)255;
        return p;
    };
    __bf16* Ks   = (__bf16*)carve((size_t)NN * NN * 2);
    __bf16* Kb   = (__bf16*)carve((size_t)NN * NN * 2);
    float*  rns  = (float*)carve(NN * 4);
    float*  rnb  = (float*)carve(NN * 4);
    __bf16* sm   = (__bf16*)carve((size_t)CP * NN * 2);
    float*  ut   = (float*)carve((size_t)CC * NN * 4);
    float*  q    = (float*)carve((size_t)CC * NN * 4);
    float*  part = (float*)carve((size_t)2 * KSPLIT * CP * NN * 4);
    float*  so   = (float*)carve((size_t)CC * NN * 4);
    float*  bo   = (float*)carve((size_t)CC * NN * 4);
    float*  Ms   = (float*)carve(CC * CC * 4);
    float*  Mb   = (float*)carve(CC * CC * 4);

    // one-time: kernel matrices, norms, combined mixing matrices, q init
    crf_build_kernels<<<dim3(NN / 16, NN / 16), 256, 0, stream>>>(rgb, Ks, Kb);
    crf_row_norm<<<NN, 256, 0, stream>>>(Ks, rns);
    crf_row_norm<<<NN, 256, 0, stream>>>(Kb, rnb);
    crf_mixmat<<<1, 512, 0, stream>>>(Sk, Bk, Cm, Ms, Mb);
    crf_init_q<<<(CC * NN + 255) / 256, 256, 0, stream>>>(unaries, ut, q);

    // mean-field iterations
    const int gemm_blocks = (NN / (16 * NSTRIP)) * KSPLIT * 2 / 8; // 100
    for (int it = 0; it < NUM_ITER; ++it) {
        crf_softmax<<<(NN + 255) / 256, 256, 0, stream>>>(q, sm);
        crf_gemm<<<gemm_blocks, 256, 0, stream>>>(sm, Ks, Kb, part);
        crf_reduce<<<(CC * NN + 255) / 256, 256, 0, stream>>>(part, rns, rnb, so, bo);
        crf_update_q<<<dim3((NN + 255) / 256, CC), 256, 0, stream>>>(ut, so, bo, Ms, Mb, q);
    }

    crf_write_out<<<(NN * CC + 255) / 256, 256, 0, stream>>>(q, out);
}
